// HybridKernelLSTM_65481071404312
// MI455X (gfx1250) — compile-verified, matrix operand order checked by reading, not run on hardware
//
#include <hip/hip_runtime.h>
#include <hip/hip_bf16.h>
#include <math.h>

typedef __attribute__((ext_vector_type(16))) _Float16 v16h;
typedef __attribute__((ext_vector_type(8)))  float    v8f;

#define SEQ   8192
#define EMB   256
#define NG    16      // 4 gates x 4 qubits
#define TAGS  10

// ---------------------------------------------------------------------------
// Kernel 1: G[t, n] = sum_k embed[sentence[t]][k] * Wcat[k][n] + bcat[n]
// One wave32 per 16x16 output tile; K=256 swept in 8 x v_wmma_f32_16x16x32_f16.
// A fragment = 16 gathered embedding rows (f32 -> f16), B fragment = the
// interleaved gate weights. C accumulates in f32 per the ISA VGPR layout.
// ---------------------------------------------------------------------------
__global__ __launch_bounds__(128)
void qlstm_gates_wmma(const int*   __restrict__ sentence,
                      const float* __restrict__ embed,
                      const float* __restrict__ Wf, const float* __restrict__ bf,
                      const float* __restrict__ Wi, const float* __restrict__ bi,
                      const float* __restrict__ Wu, const float* __restrict__ bu,
                      const float* __restrict__ Wo, const float* __restrict__ bo,
                      float* __restrict__ G)
{
    const int lane = threadIdx.x & 31;
    const int wave = threadIdx.x >> 5;
    const int tile = blockIdx.x * 4 + wave;   // 512 tiles of 16 tokens
    const int half = lane >> 4;               // K-half selector (ISA A/B layout)
    const int m    = lane & 15;               // A row / B,C column within tile

    // B column n = m -> gate = n>>2, qubit = n&3 (Wcat = [Wf|Wi|Wu|Wo][:256])
    const int gate  = m >> 2;
    const int qubit = m & 3;
    const float* Wg = (gate == 0) ? Wf : (gate == 1) ? Wi : (gate == 2) ? Wu : Wo;
    const float* bg = (gate == 0) ? bf : (gate == 1) ? bi : (gate == 2) ? bu : bo;

    const int    tok  = sentence[tile * 16 + m];
    const float* rowp = embed + (size_t)tok * EMB;

    v8f acc = {};

    #pragma unroll
    for (int kb = 0; kb < EMB; kb += 32) {
        // ---- A fragment (16-bit A 16x32 layout) ----
        // elems 0..7  hold K = kb + half*8 + i
        // elems 8..15 hold K = kb + 16 + half*8 + (i-8)
        const int k0 = kb + half * 8;
        const float4 a0 = *(const float4*)(rowp + k0);
        const float4 a1 = *(const float4*)(rowp + k0 + 4);
        const float4 a2 = *(const float4*)(rowp + k0 + 16);
        const float4 a3 = *(const float4*)(rowp + k0 + 20);
        v16h a;
        a[0]=(_Float16)a0.x;  a[1]=(_Float16)a0.y;  a[2]=(_Float16)a0.z;  a[3]=(_Float16)a0.w;
        a[4]=(_Float16)a1.x;  a[5]=(_Float16)a1.y;  a[6]=(_Float16)a1.z;  a[7]=(_Float16)a1.w;
        a[8]=(_Float16)a2.x;  a[9]=(_Float16)a2.y;  a[10]=(_Float16)a2.z; a[11]=(_Float16)a2.w;
        a[12]=(_Float16)a3.x; a[13]=(_Float16)a3.y; a[14]=(_Float16)a3.z; a[15]=(_Float16)a3.w;

        // ---- B fragment (32x16, same K striping; column = this lane's n) ----
        v16h b;
        #pragma unroll
        for (int i = 0; i < 8; ++i) {
            b[i]     = (_Float16)Wg[(k0 + i)      * 4 + qubit];
            b[i + 8] = (_Float16)Wg[(k0 + 16 + i) * 4 + qubit];
        }

        if (kb + 32 < EMB)
            __builtin_prefetch(rowp + kb + 32, 0, 0);   // global_prefetch_b8

        // v_wmma_f32_16x16x32_f16: D = A x B + C
        acc = __builtin_amdgcn_wmma_f32_16x16x32_f16(
                  false, a, false, b, (short)0, acc, false, false);
    }

    // C/D layout: VGPR v -> row (v + 8*half), column m. Fold in the gate bias.
    const float bias = bg[qubit];
    #pragma unroll
    for (int v = 0; v < 8; ++v) {
        const int row = v + half * 8;
        G[(size_t)(tile * 16 + row) * NG + m] = acc[v] + bias;
    }
}

// ---------------------------------------------------------------------------
// Kernel 2: the inherently serial LSTM scan. 16 lanes of one wave32 each own
// one (gate,qubit) column; cross-lane data flows via __shfl (no LDS traffic).
// Per step only the 4x16 hidden matmul + cos/sigmoid/tanh sit on the
// dependent chain -- everything K=256 related was hoisted into kernel 1.
// ---------------------------------------------------------------------------
__global__ void qlstm_scan(const float* __restrict__ G,
                           const float* __restrict__ Wf, const float* __restrict__ Wi,
                           const float* __restrict__ Wu, const float* __restrict__ Wo,
                           const float* __restrict__ pf, const float* __restrict__ pi_,
                           const float* __restrict__ pu, const float* __restrict__ po,
                           float* __restrict__ outs)
{
    const int l = threadIdx.x;
    if (l >= NG) return;
    const int gate  = l >> 2;
    const int qubit = l & 3;
    const int w     = l & 3;            // hidden index this lane also tracks

    const float* Wg = (gate == 0) ? Wf : (gate == 1) ? Wi : (gate == 2) ? Wu : Wo;
    const float* pg = (gate == 0) ? pf : (gate == 1) ? pi_ : (gate == 2) ? pu : po;
    const float phase = pg[qubit];
    const float whh0 = Wg[(EMB + 0) * 4 + qubit];
    const float whh1 = Wg[(EMB + 1) * 4 + qubit];
    const float whh2 = Wg[(EMB + 2) * 4 + qubit];
    const float whh3 = Wg[(EMB + 3) * 4 + qubit];

    float hx0 = 0.f, hx1 = 0.f, hx2 = 0.f, hx3 = 0.f;
    float cx = 0.f, h_mine = 0.f;

    for (int t = 0; t < SEQ; ++t) {
        float aval = G[(size_t)t * NG + l]
                   + hx0 * whh0 + hx1 * whh1 + hx2 * whh2 + hx3 * whh3;
        float z = cosf(aval + phase);

        // gather the 4 qubit cosines of this lane's gate
        const int qb = gate << 2;
        float z0 = __shfl(z, qb + 0, 32);
        float z1 = __shfl(z, qb + 1, 32);
        float z2 = __shfl(z, qb + 2, 32);
        float z3 = __shfl(z, qb + 3, 32);

        float q;
        if      (qubit == 0) q = z1 * z2 * z3;
        else if (qubit == 1) q = z0 * z1;
        else if (qubit == 2) q = z0 * z1 * z2;
        else                 q = z0 * z1 * z2 * z3;

        float act = (gate == 2) ? tanhf(q) : 1.0f / (1.0f + expf(-q));

        // gather f,i,g,o for hidden index w; lanes with equal w stay coherent
        float fv = __shfl(act,  0 + w, 32);
        float iv = __shfl(act,  4 + w, 32);
        float gv = __shfl(act,  8 + w, 32);
        float ov = __shfl(act, 12 + w, 32);

        cx     = fv * cx + iv * gv;
        h_mine = ov * tanhf(cx);

        if (l < 4) outs[(size_t)t * 4 + l] = h_mine;

        hx0 = __shfl(h_mine, 0, 32);
        hx1 = __shfl(h_mine, 1, 32);
        hx2 = __shfl(h_mine, 2, 32);
        hx3 = __shfl(h_mine, 3, 32);
    }
}

// ---------------------------------------------------------------------------
// Kernel 3: logits = outs @ W_tag + b_tag, then log_softmax. One lane/token.
// (K=4, N=10: a WMMA tile would be 75% padding; plain VALU is the right call.)
// ---------------------------------------------------------------------------
__global__ __launch_bounds__(256)
void qlstm_logits(const float* __restrict__ outs,
                  const float* __restrict__ W_tag,
                  const float* __restrict__ b_tag,
                  float* __restrict__ out)
{
    const int t = blockIdx.x * blockDim.x + threadIdx.x;
    if (t >= SEQ) return;
    const float h0 = outs[t * 4 + 0], h1 = outs[t * 4 + 1];
    const float h2 = outs[t * 4 + 2], h3 = outs[t * 4 + 3];

    float lg[TAGS];
    float mx = -INFINITY;
    #pragma unroll
    for (int j = 0; j < TAGS; ++j) {
        float v = b_tag[j]
                + h0 * W_tag[0 * TAGS + j] + h1 * W_tag[1 * TAGS + j]
                + h2 * W_tag[2 * TAGS + j] + h3 * W_tag[3 * TAGS + j];
        lg[j] = v;
        mx = fmaxf(mx, v);
    }
    float s = 0.f;
    #pragma unroll
    for (int j = 0; j < TAGS; ++j) s += expf(lg[j] - mx);
    const float lse = mx + logf(s);
    #pragma unroll
    for (int j = 0; j < TAGS; ++j) out[(size_t)t * TAGS + j] = lg[j] - lse;
}

extern "C" void kernel_launch(void* const* d_in, const int* in_sizes, int n_in,
                              void* d_out, int out_size, void* d_ws, size_t ws_size,
                              hipStream_t stream) {
    // setup_inputs() dict order
    const int*   sentence = (const int*)  d_in[0];
    const float* embed    = (const float*)d_in[1];
    const float* Wf       = (const float*)d_in[2];
    const float* bf       = (const float*)d_in[3];
    const float* Wi       = (const float*)d_in[4];
    const float* bi       = (const float*)d_in[5];
    const float* Wu       = (const float*)d_in[6];
    const float* bu       = (const float*)d_in[7];
    const float* Wo       = (const float*)d_in[8];
    const float* bo       = (const float*)d_in[9];
    const float* pf       = (const float*)d_in[10];
    const float* pi_      = (const float*)d_in[11];
    const float* pu       = (const float*)d_in[12];
    const float* po       = (const float*)d_in[13];
    const float* W_tag    = (const float*)d_in[14];
    const float* b_tag    = (const float*)d_in[15];
    float*       out      = (float*)d_out;

    // Workspace: G[8192*16] (512 KB) then outs[8192*4] (128 KB)
    float* G    = (float*)d_ws;
    float* outs = G + (size_t)SEQ * NG;

    // Phase 1: 512 tiles, 4 waves/block -> 128 blocks of 128 threads
    qlstm_gates_wmma<<<128, 128, 0, stream>>>(sentence, embed,
                                              Wf, bf, Wi, bi, Wu, bu, Wo, bo, G);
    // Phase 2: single wave32 serial scan
    qlstm_scan<<<1, 32, 0, stream>>>(G, Wf, Wi, Wu, Wo, pf, pi_, pu, po, outs);
    // Phase 3: projection + log_softmax
    qlstm_logits<<<(SEQ + 255) / 256, 256, 0, stream>>>(outs, W_tag, b_tag, out);
}
